// LTFGW_18451179504147
// MI455X (gfx1250) — matrix operand order repaired and checked by reading, not compiled, not run on hardware
//
#include <hip/hip_runtime.h>

// ---------------- problem constants (from reference) ----------------
#define NNODE   10000
#define DFEAT   128
#define TT      10            // templates
#define KK      10            // template nodes
#define MM      17            // local star-graph nodes (1 + DEG)
#define DEG     16
#define NOUTER  5
#define NSINK   10
// EPS = 0.5 -> 1/EPS = 2 ; ALPHA = 0.5

typedef __attribute__((ext_vector_type(16))) __bf16 bf16x16;
typedef __attribute__((ext_vector_type(8)))  float  f32x8;

__device__ __forceinline__ void cvt_split(float f, __bf16& h, __bf16& l) {
    h = (__bf16)f;                 // hi part (RNE)
    l = (__bf16)(f - (float)h);    // residual
}

// =====================================================================
// Kernel A: G = x @ tf^T   [10000 x 100]  via bf16 hi/lo split WMMA.
// grid = 625 blocks (16 rows each), block = 224 thr = 7 waves,
// wave w owns column tile w (16 cols, cols 100..111 are clamped/dropped).
// =====================================================================
__global__ __launch_bounds__(224) void ltfgw_gemm(const float* __restrict__ x,
                                                  const float* __restrict__ tf,
                                                  float* __restrict__ G) {
    const int tid    = threadIdx.x;
    const int wave   = tid >> 5;
    const int lane   = tid & 31;
    const int l15    = lane & 15;
    const int hiHalf = (lane >> 4) & 1;
    const int row16  = blockIdx.x * 16;

    const int col  = wave * 16 + l15;                 // output column (t*10+j)
    const int colc = (col < TT * KK) ? col : 0;       // clamp OOB reads

    f32x8 acc = {0.f, 0.f, 0.f, 0.f, 0.f, 0.f, 0.f, 0.f};

#pragma unroll
    for (int kk = 0; kk < DFEAT; kk += 32) {
        // ---- A fragment: x rows. lane<16: K{kk..kk+15};
        //                          lane>=16: K{kk+16..kk+31}
        const int ar = row16 + l15;
        const int ak = kk + (hiHalf ? 16 : 0);
        const float4* ap0 = (const float4*)(x + (size_t)ar * DFEAT + ak);
        const float4* ap1 = (const float4*)(x + (size_t)ar * DFEAT + ak + 8);
        float4 a0 = ap0[0], a1 = ap0[1];
        float4 a2 = ap1[0], a3 = ap1[1];
        float av[16] = {a0.x, a0.y, a0.z, a0.w, a1.x, a1.y, a1.z, a1.w,
                        a2.x, a2.y, a2.z, a2.w, a3.x, a3.y, a3.z, a3.w};
        bf16x16 ahi, alo;
#pragma unroll
        for (int e = 0; e < 16; ++e) { __bf16 h, l; cvt_split(av[e], h, l); ahi[e] = h; alo[e] = l; }

        // ---- B fragment: B[k, col] = tf[col, k]; lane=column,
        //      elements = 16 consecutive k (lanes>=16 start at kk+16)
        const int bk = kk + (hiHalf ? 16 : 0);
        const float4* bp = (const float4*)(tf + (size_t)colc * DFEAT + bk);
        float4 b0 = bp[0], b1 = bp[1], b2 = bp[2], b3 = bp[3];
        float bv[16] = {b0.x, b0.y, b0.z, b0.w, b1.x, b1.y, b1.z, b1.w,
                        b2.x, b2.y, b2.z, b2.w, b3.x, b3.y, b3.z, b3.w};
        bf16x16 bhi, blo;
#pragma unroll
        for (int e = 0; e < 16; ++e) { __bf16 h, l; cvt_split(bv[e], h, l); bhi[e] = h; blo[e] = l; }

        // fp32-ish product: hi*hi + hi*lo + lo*hi
        acc = __builtin_amdgcn_wmma_f32_16x16x32_bf16(false, ahi, false, bhi, (short)0, acc, false, false);
        acc = __builtin_amdgcn_wmma_f32_16x16x32_bf16(false, ahi, false, blo, (short)0, acc, false, false);
        acc = __builtin_amdgcn_wmma_f32_16x16x32_bf16(false, alo, false, bhi, (short)0, acc, false, false);
    }

    if (col < TT * KK) {
#pragma unroll
        for (int r = 0; r < 8; ++r) {
            const int mrow = r + (hiHalf ? 8 : 0);
            G[(size_t)(row16 + mrow) * (TT * KK) + col] = acc[r];
        }
    }
}

// =====================================================================
// Kernel: squared row norms (d=128), wave per row, butterfly reduce.
// =====================================================================
__global__ __launch_bounds__(256) void row_norms(const float* __restrict__ v,
                                                 float* __restrict__ out, int nrows) {
    const int w    = (blockIdx.x * blockDim.x + threadIdx.x) >> 5;
    const int lane = threadIdx.x & 31;
    const int nw   = (gridDim.x * blockDim.x) >> 5;
    for (int row = w; row < nrows; row += nw) {
        const float* p = v + (size_t)row * DFEAT;
        float s = 0.f;
#pragma unroll
        for (int c = 0; c < DFEAT; c += 32) { float t = p[c + lane]; s += t * t; }
#pragma unroll
        for (int m = 16; m; m >>= 1) s += __shfl_xor(s, m, 32);
        if (lane == 0) out[row] = s;
    }
}

// =====================================================================
// Kernel B: one wave32 per (node, template) pair; lane = local row.
// Register-resident log-domain Sinkhorn; column LSE via shfl_xor butterfly.
// =====================================================================
__global__ __launch_bounds__(256) void ltfgw_sinkhorn(const int* __restrict__ src,
                                                      const float* __restrict__ templates,
                                                      const float* __restrict__ Gm,
                                                      const float* __restrict__ xn2,
                                                      const float* __restrict__ tn2,
                                                      float* __restrict__ out) {
    __shared__ float sCt[TT * KK * KK];   // template adjacencies
    __shared__ float sCtq[TT * KK];       // (Ct^2 @ q)[j]
    __shared__ float sTn2[TT * KK];       // template feature sq-norms

    const int tid = threadIdx.x;
    for (int i = tid; i < TT * KK * KK; i += 256) sCt[i]  = templates[i];
    for (int i = tid; i < TT * KK;      i += 256) sTn2[i] = tn2[i];
    __syncthreads();
    for (int i = tid; i < TT * KK; i += 256) {
        float s = 0.f;
#pragma unroll
        for (int k2 = 0; k2 < KK; ++k2) { float c = sCt[i * KK + k2]; s += c * c; }
        sCtq[i] = s * (1.0f / KK);
    }
    __syncthreads();

    const int pair = blockIdx.x * 8 + (tid >> 5);
    if (pair >= NNODE * TT) return;                       // wave-uniform
    const int node = pair / TT;
    const int t    = pair - node * TT;
    const int lane = tid & 31;
    const int row  = lane;                                // local-graph row
    const bool act = row < MM;

    int r = node;                                         // row 0 = the node itself
    if (act && row > 0) r = src[node * DEG + row - 1];    // rows 1..16 = neighbors

    const float LOGP = -2.8332133f;    // -log(17)
    const float LOGQ = -2.3025851f;    // -log(10)
    const float cip  = (row == 0) ? (16.0f / 17.0f) : (1.0f / 17.0f);  // (Ci^2 @ p)[row]

    // feature cost row: M[j] = (|x_r|^2 + |tf_{t,j}|^2 - 2 G[r, t*K+j]) / d
    float M[KK];
    {
        const float xn = xn2[r];
        const float* gp = Gm + (size_t)r * (TT * KK) + t * KK;
#pragma unroll
        for (int j = 0; j < KK; ++j)
            M[j] = (xn + sTn2[t * KK + j] - 2.0f * gp[j]) * (1.0f / DFEAT);
    }
    const float* ct = &sCt[t * KK * KK];

    float logT[KK];
#pragma unroll
    for (int j = 0; j < KK; ++j) logT[j] = LOGP + LOGQ;   // log(p_i q_j)
    float logK[KK], g[KK], f = 0.0f;

    for (int o = 0; o < NOUTER; ++o) {
        // Tp = exp(logT); star-graph GW product needs Tp[0,:] and colsum(rows 1..16)
        float Tp[KK], v[KK];
#pragma unroll
        for (int j = 0; j < KK; ++j) Tp[j] = __expf(logT[j]);
#pragma unroll
        for (int j = 0; j < KK; ++j) {
            float c = act ? Tp[j] : 0.0f;
#pragma unroll
            for (int m = 16; m; m >>= 1) c += __shfl_xor(c, m, 32);
            const float tp0 = __int_as_float(__builtin_amdgcn_readfirstlane(__float_as_int(Tp[j])));
            v[j] = (row == 0) ? (c - tp0) : tp0;          // (Ci@Tp)[row, :]
        }
        // cost row + mirror step; constC = cip + ctq[j]
#pragma unroll
        for (int j = 0; j < KK; ++j) {
            float dot = 0.0f;
#pragma unroll
            for (int k2 = 0; k2 < KK; ++k2) dot = __fmaf_rn(v[k2], ct[j * KK + k2], dot);
            const float cost = 0.5f * M[j] + (cip + sCtq[t * KK + j]) - 2.0f * dot;
            logK[j] = logT[j] - 2.0f * cost;              // - cost/EPS, EPS = 0.5
            g[j] = 0.0f;
        }
        // Sinkhorn projections
        for (int it = 0; it < NSINK; ++it) {
            float s = 0.0f;
#pragma unroll
            for (int j = 0; j < KK; ++j) s += __expf(logK[j] + g[j]);
            f = LOGP - __logf(s);                         // row scaling
#pragma unroll
            for (int j = 0; j < KK; ++j) {                // column scaling (cross-lane)
                float e = act ? __expf(logK[j] + f) : 0.0f;
#pragma unroll
                for (int m = 16; m; m >>= 1) e += __shfl_xor(e, m, 32);
                g[j] = LOGQ - __logf(e);
            }
        }
#pragma unroll
        for (int j = 0; j < KK; ++j) logT[j] = logK[j] + f + g[j];
    }

    // objective: sum Tp * ((1-a)M + a*(constC - 2*Ci@Tp@Ct^T)), a = 0.5
    float Tp[KK], v[KK];
#pragma unroll
    for (int j = 0; j < KK; ++j) Tp[j] = __expf(logT[j]);
#pragma unroll
    for (int j = 0; j < KK; ++j) {
        float c = act ? Tp[j] : 0.0f;
#pragma unroll
        for (int m = 16; m; m >>= 1) c += __shfl_xor(c, m, 32);
        const float tp0 = __int_as_float(__builtin_amdgcn_readfirstlane(__float_as_int(Tp[j])));
        v[j] = (row == 0) ? (c - tp0) : tp0;
    }
    float obj = 0.0f;
#pragma unroll
    for (int j = 0; j < KK; ++j) {
        float dot = 0.0f;
#pragma unroll
        for (int k2 = 0; k2 < KK; ++k2) dot = __fmaf_rn(v[k2], ct[j * KK + k2], dot);
        const float gt = cip + sCtq[t * KK + j] - 2.0f * dot;
        obj += Tp[j] * (0.5f * M[j] + 0.5f * gt);
    }
    obj = act ? obj : 0.0f;
#pragma unroll
    for (int m = 16; m; m >>= 1) obj += __shfl_xor(obj, m, 32);
    if (lane == 0) out[pair] = obj;
}

// =====================================================================
extern "C" void kernel_launch(void* const* d_in, const int* in_sizes, int n_in,
                              void* d_out, int out_size, void* d_ws, size_t ws_size,
                              hipStream_t stream) {
    const float* x   = (const float*)d_in[0];   // [10000, 128]
    const int*   ei  = (const int*)d_in[1];     // [2, 160000]; row 0 = src
    const float* tpl = (const float*)d_in[2];   // [10, 10, 10]
    const float* tf  = (const float*)d_in[3];   // [10, 10, 128]

    float* G    = (float*)d_ws;                       // [10000 x 100]   4.0 MB
    float* xn2  = G + (size_t)NNODE * TT * KK;        // [10000]
    float* tn2  = xn2 + NNODE;                        // [100]
    float* out  = (float*)d_out;                      // [10000 x 10]

    ltfgw_gemm    <<<dim3(NNODE / 16),     dim3(224), 0, stream>>>(x, tf, G);
    row_norms     <<<dim3(64),             dim3(256), 0, stream>>>(x, xn2, NNODE);
    row_norms     <<<dim3(4),              dim3(256), 0, stream>>>(tf, tn2, TT * KK);
    ltfgw_sinkhorn<<<dim3(NNODE * TT / 8), dim3(256), 0, stream>>>(ei, tpl, G, xn2, tn2, out);
}
